// SoftmaxGNN_81681688035484
// MI455X (gfx1250) — matrix operand order repaired and checked by reading, not compiled, not run on hardware
//
#include <hip/hip_runtime.h>
#include <hip/hip_fp16.h>

#define NNODES 100000
#define NEDGES 1000000
#define NDF 64
#define EDF 64
#define MH  128
#define MD  64
#define GH  64
#define UH  128
#define OD  64
#define FIN 192   // EDF + 2*NDF

typedef __attribute__((ext_vector_type(16))) _Float16 v16h;
typedef __attribute__((ext_vector_type(8)))  float    v8f;
typedef __attribute__((ext_vector_type(4)))  float    f4;

union AF { v16h v; unsigned u[8]; };
union BF { v16h v; f4 f[2]; };

__device__ __forceinline__ v8f wmma16(v16h a, v16h b, v8f c) {
  return __builtin_amdgcn_wmma_f32_16x16x32_f16(false, a, false, b, (short)0, c, false, false);
}

// A-fragment (16x32 f16) from LDS tile stored row-major [16][W] halves.
// ISA layout: lanes 0-15 row M=lane, K=0..7 (V0-3) & 16..23 (V4-7);
// lanes 16-31 same rows, K=8..15 & 24..31.
__device__ __forceinline__ v16h loadA_lds(const __half* tile, int W, int kt, int lane) {
  const int row = lane & 15;
  const int kh  = (lane >> 4) << 3;           // +8 K for upper half-wave
  const __half* p = tile + row * W + kt * 32 + kh;
  AF a;
  a.u[0] = *(const unsigned*)(p + 0);
  a.u[1] = *(const unsigned*)(p + 2);
  a.u[2] = *(const unsigned*)(p + 4);
  a.u[3] = *(const unsigned*)(p + 6);
  a.u[4] = *(const unsigned*)(p + 16);
  a.u[5] = *(const unsigned*)(p + 18);
  a.u[6] = *(const unsigned*)(p + 20);
  a.u[7] = *(const unsigned*)(p + 22);
  return a.v;
}

// B-fragment pre-packed in workspace: frag -> 32 lanes x 16 halves (32B/lane).
__device__ __forceinline__ v16h loadB_frag(const __half* wbuf, int frag, int lane) {
  const f4* p = (const f4*)(wbuf + ((size_t)frag * 32 + lane) * 16);
  BF b; b.f[0] = p[0]; b.f[1] = p[1];
  return b.v;
}

// monotone float <-> int encoding for integer atomicMax
__device__ __forceinline__ int   enc_f(float f) { int i = __float_as_int(f); return (i >= 0) ? i : (i ^ 0x7FFFFFFF); }
__device__ __forceinline__ float dec_f(int k)   { return __int_as_float((k >= 0) ? k : (k ^ 0x7FFFFFFF)); }

// ---- repack fp32 row-major W[K][N] into WMMA B-fragments (f16) ----
__global__ void repack_kernel(const float* __restrict__ W, int Ntot, __half* __restrict__ out) {
  const int f  = blockIdx.x;
  const int NT = Ntot >> 4;
  const int kt = f / NT, nt = f % NT;
  const int lane = threadIdx.x;
  const int n  = lane & 15;
  const int kb = (lane >> 4) << 3;            // lanes 16-31 hold K+8 (K=8..15 & 24..31)
  __half* o = out + ((size_t)f * 32 + lane) * 16;
  for (int j = 0; j < 8; ++j)
    for (int p = 0; p < 2; ++p) {
      const int k = kt * 32 + kb + 2 * j + p + (j >= 4 ? 8 : 0);
      o[j * 2 + p] = __float2half(W[(size_t)k * Ntot + nt * 16 + n]);
    }
}

__global__ void init_kernel(int* __restrict__ segmax, float* __restrict__ segsum,
                            float* __restrict__ naggr) {
  const int i = blockIdx.x * blockDim.x + threadIdx.x;
  if (i < NNODES * MD) naggr[i] = 0.f;
  if (i < NNODES) { segmax[i] = (int)0x807FFFFFu; segsum[i] = 0.f; }  // enc(-inf)
}

// ---- edge pipeline: message MLP + gate logits + segment max ----
__global__ __launch_bounds__(128) void edge_kernel(
    const float* __restrict__ nodef, const float* __restrict__ edgef,
    const int* __restrict__ sidx, const int* __restrict__ ridx,
    const float* __restrict__ bm1, const float* __restrict__ bm2,
    const float* __restrict__ bg1, const float* __restrict__ wg2, const float* __restrict__ bg2,
    const __half* __restrict__ fm1, const __half* __restrict__ fm2, const __half* __restrict__ fg1,
    float* __restrict__ emsg, float* __restrict__ elogit, int* __restrict__ segmax)
{
  __shared__ __half lds[4 * 5120];
  const int lane = threadIdx.x & 31;
  const int wave = threadIdx.x >> 5;
  __half* wl    = lds + wave * 5120;
  __half* in_t  = wl;          // 16 x 192 input tile
  __half* msg_t = wl;          // 16 x 64  (overlays in_t after layer1)
  __half* g_t   = wl + 1024;   // 16 x 64  (overlays in_t)
  __half* h_t   = wl + 3072;   // 16 x 128 hidden

  const int e0 = (blockIdx.x * 4 + wave) * 16;

  // stage concat(edge_feat, sender_feat, recv_feat) as f16
  for (int r = 0; r < 16; ++r) {
    const int e = e0 + r;
    const int s = sidx[e];
    const int rv = ridx[e];
    for (int c = lane; c < FIN; c += 32) {
      float v;
      if (c < EDF)            v = edgef[(size_t)e  * EDF + c];
      else if (c < EDF + NDF) v = nodef[(size_t)s  * NDF + (c - EDF)];
      else                    v = nodef[(size_t)rv * NDF + (c - EDF - NDF)];
      in_t[r * FIN + c] = __float2half(v);
    }
  }
  __syncthreads();

  const int col  = lane & 15;
  const int rofs = (lane >> 4) << 3;   // D layout: upper half-wave holds rows M+8

  // message layer 1: [16,192]@[192,128], relu -> h_t
  for (int nt = 0; nt < MH / 16; ++nt) {
    v8f acc = {};
    for (int kt = 0; kt < FIN / 32; ++kt)
      acc = wmma16(loadA_lds(in_t, FIN, kt, lane), loadB_frag(fm1, kt * (MH / 16) + nt, lane), acc);
    const float bias = bm1[nt * 16 + col];
    for (int v = 0; v < 8; ++v) {
      float x = acc[v] + bias; x = x > 0.f ? x : 0.f;
      h_t[(v + rofs) * MH + nt * 16 + col] = __float2half(x);
    }
  }
  __syncthreads();

  // message layer 2: [16,128]@[128,64] -> e_msg (global + LDS f16)
  for (int nt = 0; nt < MD / 16; ++nt) {
    v8f acc = {};
    for (int kt = 0; kt < MH / 32; ++kt)
      acc = wmma16(loadA_lds(h_t, MH, kt, lane), loadB_frag(fm2, kt * (MD / 16) + nt, lane), acc);
    const float bias = bm2[nt * 16 + col];
    for (int v = 0; v < 8; ++v) {
      const float x = acc[v] + bias;
      const int row = v + rofs;
      emsg[(size_t)(e0 + row) * MD + nt * 16 + col] = x;
      msg_t[row * MD + nt * 16 + col] = __float2half(x);
    }
  }
  __syncthreads();

  // gate layer 1: [16,64]@[64,64], relu -> g_t
  for (int nt = 0; nt < GH / 16; ++nt) {
    v8f acc = {};
    for (int kt = 0; kt < MD / 32; ++kt)
      acc = wmma16(loadA_lds(msg_t, MD, kt, lane), loadB_frag(fg1, kt * (GH / 16) + nt, lane), acc);
    const float bias = bg1[nt * 16 + col];
    for (int v = 0; v < 8; ++v) {
      float x = acc[v] + bias; x = x > 0.f ? x : 0.f;
      g_t[(v + rofs) * GH + nt * 16 + col] = __float2half(x);
    }
  }
  __syncthreads();

  // gate layer 2: dot(g, wg2) -> logit; segment max via ordered-int atomicMax
  {
    const int row = lane & 15;
    const int cb  = (lane >> 4) * 32;
    float s = 0.f;
    for (int j = 0; j < 32; ++j)
      s += __half2float(g_t[row * GH + cb + j]) * wg2[cb + j];
    s += __shfl_xor(s, 16, 32);
    if (lane < 16) {
      const int e = e0 + row;
      const float logit = s + bg2[0];
      elogit[e] = logit;
      atomicMax(&segmax[ridx[e]], enc_f(logit));
    }
  }
}

__global__ void expsum_kernel(const float* __restrict__ elogit, const int* __restrict__ ridx,
                              const int* __restrict__ segmax, float* __restrict__ segsum) {
  const int e = blockIdx.x * blockDim.x + threadIdx.x;
  if (e >= NEDGES) return;
  const int rv = ridx[e];
  atomicAdd(&segsum[rv], expf(elogit[e] - dec_f(segmax[rv])));
}

__global__ void scatter_kernel(const float* __restrict__ elogit, const int* __restrict__ ridx,
                               const int* __restrict__ segmax, const float* __restrict__ segsum,
                               const float* __restrict__ emsg, float* __restrict__ naggr) {
  const int t = blockIdx.x * blockDim.x + threadIdx.x;
  if (t >= NEDGES * 4) return;
  const int e = t >> 2, q = t & 3;
  const int rv = ridx[e];
  const float w = expf(elogit[e] - dec_f(segmax[rv])) / segsum[rv];
  const f4* m = (const f4*)(emsg + (size_t)e * MD + q * 16);
  float* dst = naggr + (size_t)rv * MD + q * 16;
  for (int i = 0; i < 4; ++i) {
    f4 v = m[i];
    atomicAdd(dst + i * 4 + 0, w * v.x);
    atomicAdd(dst + i * 4 + 1, w * v.y);
    atomicAdd(dst + i * 4 + 2, w * v.z);
    atomicAdd(dst + i * 4 + 3, w * v.w);
  }
}

// ---- node update MLP ----
__global__ __launch_bounds__(64) void node_kernel(
    const float* __restrict__ nodef, const float* __restrict__ naggr,
    const float* __restrict__ bu1, const float* __restrict__ bu2,
    const __half* __restrict__ fu1, const __half* __restrict__ fu2,
    float* __restrict__ nout)
{
  __shared__ __half lds[2 * 4096];
  const int lane = threadIdx.x & 31;
  const int wave = threadIdx.x >> 5;
  __half* wl  = lds + wave * 4096;
  __half* u_t = wl;          // 16 x 128
  __half* h_t = wl + 2048;   // 16 x 128

  const int n0 = (blockIdx.x * 2 + wave) * 16;
  const int UIN = NDF + MD;  // 128

  for (int r = 0; r < 16; ++r) {
    const int n = n0 + r;
    for (int c = lane; c < UIN; c += 32) {
      const float v = (c < NDF) ? nodef[(size_t)n * NDF + c] : naggr[(size_t)n * MD + (c - NDF)];
      u_t[r * UIN + c] = __float2half(v);
    }
  }
  __syncthreads();

  const int col  = lane & 15;
  const int rofs = (lane >> 4) << 3;

  for (int nt = 0; nt < UH / 16; ++nt) {
    v8f acc = {};
    for (int kt = 0; kt < UIN / 32; ++kt)
      acc = wmma16(loadA_lds(u_t, UIN, kt, lane), loadB_frag(fu1, kt * (UH / 16) + nt, lane), acc);
    const float bias = bu1[nt * 16 + col];
    for (int v = 0; v < 8; ++v) {
      float x = acc[v] + bias; x = x > 0.f ? x : 0.f;
      h_t[(v + rofs) * UH + nt * 16 + col] = __float2half(x);
    }
  }
  __syncthreads();

  for (int nt = 0; nt < OD / 16; ++nt) {
    v8f acc = {};
    for (int kt = 0; kt < UH / 32; ++kt)
      acc = wmma16(loadA_lds(h_t, UH, kt, lane), loadB_frag(fu2, kt * (OD / 16) + nt, lane), acc);
    const float bias = bu2[nt * 16 + col];
    for (int v = 0; v < 8; ++v)
      nout[(size_t)(n0 + v + rofs) * OD + nt * 16 + col] = acc[v] + bias;
  }
}

extern "C" void kernel_launch(void* const* d_in, const int* in_sizes, int n_in,
                              void* d_out, int out_size, void* d_ws, size_t ws_size,
                              hipStream_t stream) {
  const float* nodef = (const float*)d_in[0];
  const float* edgef = (const float*)d_in[1];
  const int*   sidx  = (const int*)d_in[2];
  const int*   ridx  = (const int*)d_in[3];
  const float* Wm1 = (const float*)d_in[4];
  const float* bm1 = (const float*)d_in[5];
  const float* Wm2 = (const float*)d_in[6];
  const float* bm2 = (const float*)d_in[7];
  const float* Wg1 = (const float*)d_in[8];
  const float* bg1 = (const float*)d_in[9];
  const float* Wg2 = (const float*)d_in[10];
  const float* bg2 = (const float*)d_in[11];
  const float* Wu1 = (const float*)d_in[12];
  const float* bu1 = (const float*)d_in[13];
  const float* Wu2 = (const float*)d_in[14];
  const float* bu2 = (const float*)d_in[15];

  float* nout = (float*)d_out;                               // [N, OD]
  float* emsg = (float*)d_out + (size_t)NNODES * OD;         // [E, MD]

  uint8_t* ws = (uint8_t*)d_ws;
  size_t off = 0;
  __half* fm1 = (__half*)(ws + off); off += (size_t)48 * 512 * 2;   // 6x8 frags
  __half* fm2 = (__half*)(ws + off); off += (size_t)16 * 512 * 2;   // 4x4
  __half* fg1 = (__half*)(ws + off); off += (size_t)8  * 512 * 2;   // 2x4
  __half* fu1 = (__half*)(ws + off); off += (size_t)32 * 512 * 2;   // 4x8
  __half* fu2 = (__half*)(ws + off); off += (size_t)16 * 512 * 2;   // 4x4
  off = (off + 255) & ~(size_t)255;
  int*   segmax = (int*)(ws + off);   off += (size_t)NNODES * 4; off = (off + 255) & ~(size_t)255;
  float* segsum = (float*)(ws + off); off += (size_t)NNODES * 4; off = (off + 255) & ~(size_t)255;
  float* elogit = (float*)(ws + off); off += (size_t)NEDGES * 4; off = (off + 255) & ~(size_t)255;
  float* naggr  = (float*)(ws + off); off += (size_t)NNODES * MD * 4;

  repack_kernel<<<48, 32, 0, stream>>>(Wm1, MH, fm1);
  repack_kernel<<<16, 32, 0, stream>>>(Wm2, MD, fm2);
  repack_kernel<<<8,  32, 0, stream>>>(Wg1, GH, fg1);
  repack_kernel<<<32, 32, 0, stream>>>(Wu1, UH, fu1);
  repack_kernel<<<16, 32, 0, stream>>>(Wu2, OD, fu2);

  init_kernel<<<(NNODES * MD + 255) / 256, 256, 0, stream>>>(segmax, segsum, naggr);

  edge_kernel<<<NEDGES / 64, 128, 0, stream>>>(nodef, edgef, sidx, ridx,
                                               bm1, bm2, bg1, Wg2, bg2,
                                               fm1, fm2, fg1, emsg, elogit, segmax);

  expsum_kernel<<<(NEDGES + 255) / 256, 256, 0, stream>>>(elogit, ridx, segmax, segsum);

  scatter_kernel<<<(NEDGES * 4 + 255) / 256, 256, 0, stream>>>(elogit, ridx, segmax, segsum,
                                                               emsg, naggr);

  node_kernel<<<NNODES / 32, 64, 0, stream>>>(nodef, naggr, bu1, bu2, fu1, fu2, nout);
}
